// DTrans_50199577755710
// MI455X (gfx1250) — matrix operand order repaired and checked by reading, not compiled, not run on hardware
//
#include <hip/hip_runtime.h>
#include <hip/hip_bf16.h>

// ---------------- model constants ----------------
#define NN    1024
#define DD    2048
#define HH    8
#define DH    256      // DD/HH
#define LL    3
#define KTOP  8
#define MM    16
#define DFFC  2048
#define NC    35       // NCLS-1
#define DEC1  1024

// ---------------- WMMA types ----------------
typedef __attribute__((ext_vector_type(16))) __bf16 v16bf;
typedef __attribute__((ext_vector_type(8)))  float  v8f;

union FragBF { uint4 q[2]; v16bf v; };
union AccF   { v8f v; float f[8]; };

// pack two f32 into packed bf16 pair: +0x8000 round bias, then one v_perm_b32
__device__ __forceinline__ unsigned pk2bf(float x, float y) {
  unsigned ux = __float_as_uint(x) + 0x8000u;
  unsigned uy = __float_as_uint(y) + 0x8000u;
  // D = {uy[31:16], ux[31:16]} : sel bytes 2,3 from S1(=ux), 6,7 from S0(=uy)
  return __builtin_amdgcn_perm(uy, ux, 0x07060302u);
}

// ---------------- optional CDNA5 async global->LDS copy (ASYNCcnt path) ----------------
#if defined(__has_builtin)
#if __has_builtin(__builtin_amdgcn_global_load_async_to_lds_b128)
#define HAVE_ASYNC_LDS 1
#endif
#endif

#ifdef HAVE_ASYNC_LDS
typedef __attribute__((ext_vector_type(4))) int i4v;
typedef __attribute__((address_space(1))) i4v as1_i4;
typedef __attribute__((address_space(3))) i4v as3_i4;
#define ASYNC_EN 1
__device__ __forceinline__ void async_cp16(const float* g, float* l) {
  __builtin_amdgcn_global_load_async_to_lds_b128((as1_i4*)g, (as3_i4*)l, 0, 0);
}
__device__ __forceinline__ void wait_async0() {
#if __has_builtin(__builtin_amdgcn_s_wait_asynccnt)
  __builtin_amdgcn_s_wait_asynccnt(0);
#else
  asm volatile("s_wait_asynccnt 0x0" ::: "memory");
#endif
}
#else
#define ASYNC_EN 0
__device__ __forceinline__ void async_cp16(const float* g, float* l) { (void)g; (void)l; }
__device__ __forceinline__ void wait_async0() {}
#endif

// ---------------- generic batched GEMM: C = act(alpha * A x B + bias) ----------------
// A: [M x K] row-major (f32), lda
// B: bT==0 -> [K x N] row-major; bT==1 -> [N x K] row-major (i.e. B = Bt^T)
// Block tile 128x128, K step 32, 8 waves (wave32), wave tile 32x64 (2x4 wmma tiles).
// Double-buffered f32 LDS tiles; bf16 conversion fused into fragment build (v_perm pack).
// Full tiles of A (and B when bT==1) are moved with GLOBAL_LOAD_ASYNC_TO_LDS_B128 when
// the toolchain exposes the builtin; otherwise a register-staged synchronous fill is used.
#define BM 128
#define BN 128
#define BK 32
#define KS 36   // padded LDS stride (floats): 36*4B = 144B, multiple of 16B

__global__ __launch_bounds__(256)
void gemm_bf16_wmma(const float* __restrict__ A, long long lda, long long sA,
                    const float* __restrict__ B, long long ldb, long long sB,
                    float*       __restrict__ C, long long ldc, long long sC,
                    int Mrows, int Ncols, int Ktot,
                    const float* __restrict__ bias, float alpha, int bT, int relu)
{
  __shared__ float As[2][BM][KS];
  __shared__ float Bs[2][BN][KS];   // stored [n][k] for contiguous-K fragments

  const int bz = blockIdx.z;
  A += (long long)bz * sA;
  B += (long long)bz * sB;
  C += (long long)bz * sC;

  const int row0 = blockIdx.y * BM;
  const int col0 = blockIdx.x * BN;
  const int tid  = threadIdx.x;
  const int wid  = tid >> 5;
  const int lane = tid & 31;
  const int wm = (wid & 3) * 32;      // wave row offset inside block tile
  const int wn = (wid >> 2) * 64;     // wave col offset inside block tile
  const int lr = lane & 15;
  const int lh = lane >> 4;

  const bool asyA = ASYNC_EN && (row0 + BM <= Mrows);
  const bool asyB = ASYNC_EN && bT && (col0 + BN <= Ncols);

  AccF acc[2][4];
#pragma unroll
  for (int tm = 0; tm < 2; ++tm)
#pragma unroll
    for (int tn = 0; tn < 4; ++tn)
#pragma unroll
      for (int j = 0; j < 8; ++j) acc[tm][tn].f[j] = 0.0f;

  float4 ra[4], rb[4];   // register staging for the synchronous fill path

  // ---- async issue: per-lane 16B global->LDS copies (ASYNCcnt) ----
  auto issueA = [&](int k0, int buf) {
#pragma unroll
    for (int i = 0; i < 4; ++i) {
      int idx = tid + i * 256;
      int r   = idx >> 3;
      int kq  = (idx & 7) * 4;
      async_cp16(A + (long long)(row0 + r) * lda + (k0 + kq), &As[buf][r][kq]);
    }
  };
  auto issueB = [&](int k0, int buf) {   // bT==1 only
#pragma unroll
    for (int i = 0; i < 4; ++i) {
      int idx = tid + i * 256;
      int n   = idx >> 3;
      int kq  = (idx & 7) * 4;
      async_cp16(B + (long long)(col0 + n) * ldb + (k0 + kq), &Bs[buf][n][kq]);
    }
  };

  // ---- synchronous fill: stage into registers ----
  auto loadA_regs = [&](int k0) {
#pragma unroll
    for (int i = 0; i < 4; ++i) {
      int idx = tid + i * 256;
      int r   = idx >> 3;
      int kq  = (idx & 7) * 4;
      ra[i] = make_float4(0.f, 0.f, 0.f, 0.f);
      int gr = row0 + r;
      if (gr < Mrows)
        ra[i] = *(const float4*)(A + (long long)gr * lda + (k0 + kq));
    }
  };
  auto loadB_regs = [&](int k0) {
    if (bT) {
#pragma unroll
      for (int i = 0; i < 4; ++i) {
        int idx = tid + i * 256;
        int n   = idx >> 3;
        int kq  = (idx & 7) * 4;
        rb[i] = make_float4(0.f, 0.f, 0.f, 0.f);
        int gn = col0 + n;
        if (gn < Ncols)
          rb[i] = *(const float4*)(B + (long long)gn * ldb + (k0 + kq));
      }
    } else {
#pragma unroll
      for (int i = 0; i < 4; ++i) {
        int idx = tid + i * 256;          // 32 k rows x 32 n-quads
        int kk  = idx >> 5;
        int nq  = (idx & 31) * 4;
        rb[i] = make_float4(0.f, 0.f, 0.f, 0.f);
        int gn = col0 + nq;
        long long rowb = (long long)(k0 + kk) * ldb;
        if (((ldb & 3) == 0) && (gn + 3 < Ncols)) {
          rb[i] = *(const float4*)(B + rowb + gn);
        } else {
          if (gn + 0 < Ncols) rb[i].x = B[rowb + gn + 0];
          if (gn + 1 < Ncols) rb[i].y = B[rowb + gn + 1];
          if (gn + 2 < Ncols) rb[i].z = B[rowb + gn + 2];
          if (gn + 3 < Ncols) rb[i].w = B[rowb + gn + 3];
        }
      }
    }
  };
  auto storeA = [&](int buf) {
#pragma unroll
    for (int i = 0; i < 4; ++i) {
      int idx = tid + i * 256;
      int r   = idx >> 3;
      int kq  = (idx & 7) * 4;
      *(float4*)&As[buf][r][kq] = ra[i];
    }
  };
  auto storeB = [&](int buf) {
    if (bT) {
#pragma unroll
      for (int i = 0; i < 4; ++i) {
        int idx = tid + i * 256;
        int n   = idx >> 3;
        int kq  = (idx & 7) * 4;
        *(float4*)&Bs[buf][n][kq] = rb[i];
      }
    } else {
#pragma unroll
      for (int i = 0; i < 4; ++i) {
        int idx = tid + i * 256;
        int kk  = idx >> 5;
        int nq  = (idx & 31) * 4;
        Bs[buf][nq + 0][kk] = rb[i].x;
        Bs[buf][nq + 1][kk] = rb[i].y;
        Bs[buf][nq + 2][kk] = rb[i].z;
        Bs[buf][nq + 3][kk] = rb[i].w;
      }
    }
  };

  // ---- fragment build (f32 LDS -> bf16 regs) + 8 WMMAs ----
  auto compute = [&](int buf) {
    FragBF a[2], b[4];
#pragma unroll
    for (int t = 0; t < 2; ++t) {
      // A 16x32: lane (row lr, half lh): dw0-3 = K 8h+{0..7}, dw4-7 = K 16+8h+{0..7}
      const float* p = &As[buf][wm + t * 16 + lr][lh * 8];
      float4 x0 = *(const float4*)(p + 0);
      float4 x1 = *(const float4*)(p + 4);
      float4 x2 = *(const float4*)(p + 16);
      float4 x3 = *(const float4*)(p + 20);
      a[t].q[0] = make_uint4(pk2bf(x0.x, x0.y), pk2bf(x0.z, x0.w),
                             pk2bf(x1.x, x1.y), pk2bf(x1.z, x1.w));
      a[t].q[1] = make_uint4(pk2bf(x2.x, x2.y), pk2bf(x2.z, x2.w),
                             pk2bf(x3.x, x3.y), pk2bf(x3.z, x3.w));
    }
#pragma unroll
    for (int t = 0; t < 4; ++t) {
      // B 32x16: lane (col lr, half lh): dw0-3 = K 8h+{0..7}, dw4-7 = K 16+8h+{0..7}
      const float* p = &Bs[buf][wn + t * 16 + lr][lh * 8];
      float4 x0 = *(const float4*)(p + 0);
      float4 x1 = *(const float4*)(p + 4);
      float4 x2 = *(const float4*)(p + 16);
      float4 x3 = *(const float4*)(p + 20);
      b[t].q[0] = make_uint4(pk2bf(x0.x, x0.y), pk2bf(x0.z, x0.w),
                             pk2bf(x1.x, x1.y), pk2bf(x1.z, x1.w));
      b[t].q[1] = make_uint4(pk2bf(x2.x, x2.y), pk2bf(x2.z, x2.w),
                             pk2bf(x3.x, x3.y), pk2bf(x3.z, x3.w));
    }
#pragma unroll
    for (int tm = 0; tm < 2; ++tm)
#pragma unroll
      for (int tn = 0; tn < 4; ++tn)
        acc[tm][tn].v = __builtin_amdgcn_wmma_f32_16x16x32_bf16(
            false, a[tm].v, false, b[tn].v, (short)0, acc[tm][tn].v, false, false);
  };

  // ---- software-pipelined main loop ----
  const int nsteps = (Ktot + BK - 1) / BK;
  if (asyA) issueA(0, 0); else { loadA_regs(0); storeA(0); }
  if (asyB) issueB(0, 0); else { loadB_regs(0); storeB(0); }
  if (asyA || asyB) wait_async0();
  __syncthreads();

  for (int s = 0; s < nsteps; ++s) {
    const int buf = s & 1;
    const bool has_next = (s + 1) < nsteps;
    if (has_next) {
      const int k1 = (s + 1) * BK;
      if (asyA) issueA(k1, buf ^ 1); else loadA_regs(k1);   // overlaps WMMAs below
      if (asyB) issueB(k1, buf ^ 1); else loadB_regs(k1);
      // warm L2 for the K+2 tile (global_prefetch_b8)
      int kpf = k1 + BK;
      if (kpf < Ktot) {
        int gr = row0 + (tid >> 1);
        if (gr < Mrows)
          __builtin_prefetch((const void*)(A + (long long)gr * lda + kpf), 0, 0);
      }
    }
    compute(buf);
    if (has_next) {
      if (!asyA) storeA(buf ^ 1);
      if (!asyB) storeB(buf ^ 1);
      if (asyA || asyB) wait_async0();
    }
    __syncthreads();
  }

  // ---- epilogue: C layout = VGPR r -> M=r (lanes 0-15) / M=r+8 (lanes 16-31), N=lane%16 ----
#pragma unroll
  for (int tm = 0; tm < 2; ++tm) {
#pragma unroll
    for (int tn = 0; tn < 4; ++tn) {
      int ng = col0 + wn + tn * 16 + lr;
      if (ng >= Ncols) continue;
      float bv = (bias != nullptr) ? bias[ng] : 0.0f;
#pragma unroll
      for (int j = 0; j < 8; ++j) {
        int mg = row0 + wm + tm * 16 + j + 8 * lh;
        if (mg < Mrows) {
          float v = acc[tm][tn].f[j] * alpha + bv;
          if (relu) v = fmaxf(v, 0.0f);
          C[(long long)mg * ldc + ng] = v;
        }
      }
    }
  }
}

// ---------------- positional encoding on the fly ----------------
__device__ __forceinline__ float pe_val(int p, int d) {
  int i = d >> 1;
  // div = exp(2i * (-ln(10000)/D))
  float div = __expf((float)(2 * i) * (-9.210340371976184f / (float)DD));
  float ang = (float)p * div;
  return (d & 1) ? __cosf(ang) : __sinf(ang);
}

__device__ __forceinline__ float block_sum256(float x, float* sm) {
  int tid = threadIdx.x;
  sm[tid] = x; __syncthreads();
  for (int s = 128; s > 0; s >>= 1) {
    if (tid < s) sm[tid] += sm[tid + s];
    __syncthreads();
  }
  float r = sm[0]; __syncthreads();
  return r;
}

__device__ __forceinline__ float block_max256(float x, float* sm) {
  int tid = threadIdx.x;
  sm[tid] = x; __syncthreads();
  for (int s = 128; s > 0; s >>= 1) {
    if (tid < s) sm[tid] = fmaxf(sm[tid], sm[tid + s]);
    __syncthreads();
  }
  float r = sm[0]; __syncthreads();
  return r;
}

// ---------------- selector: top-k match attention -> s_feat [N, D] ----------------
__global__ __launch_bounds__(256)
void selector_kernel(const float* __restrict__ features,
                     const int*   __restrict__ frame_id,
                     const int*   __restrict__ match_node,
                     float*       __restrict__ s_feat)
{
  __shared__ float sred[256];
  __shared__ float sc[MM];
  __shared__ int   mn_s[MM];
  __shared__ int   pos_s[MM];
  __shared__ float w_s[KTOP];
  __shared__ int   i_s[KTOP];

  const int n   = blockIdx.x;
  const int tid = threadIdx.x;

  if (tid < MM) mn_s[tid] = match_node[n * MM + tid];
  __syncthreads();
  if (tid == 0) {
    int prev = frame_id[mn_s[0]];
    int p = 0;
    pos_s[0] = 0;
    for (int m = 1; m < MM; ++m) {
      int f = frame_id[mn_s[m]];
      if (f != prev) ++p;
      pos_s[m] = p;
      prev = f;
    }
  }
  __syncthreads();

  // 16 partial dot products, strided over D
  float part[MM];
#pragma unroll
  for (int m = 0; m < MM; ++m) part[m] = 0.0f;
  for (int d = tid; d < DD; d += 256) {
    float fv = features[(long long)n * DD + d];
#pragma unroll
    for (int m = 0; m < MM; ++m) {
      float sv = features[(long long)mn_s[m] * DD + d] + pe_val(pos_s[m], d);
      part[m] += fv * sv;
    }
  }
#pragma unroll
  for (int m = 0; m < MM; ++m) {
    float s = block_sum256(part[m], sred);
    if (tid == 0) sc[m] = s * 0.022097086912079608f;  // 1/sqrt(2048)
  }
  __syncthreads();

  if (tid == 0) {
    float tmp[MM];
    for (int m = 0; m < MM; ++m) tmp[m] = sc[m];
    float mx = -3.4e38f;
    for (int k = 0; k < KTOP; ++k) {
      int bi = 0; float bvv = -3.4e38f;
      for (int m = 0; m < MM; ++m)
        if (tmp[m] > bvv) { bvv = tmp[m]; bi = m; }
      i_s[k] = bi; w_s[k] = bvv; tmp[bi] = -3.4e38f;
      if (k == 0) mx = bvv;
    }
    float s = 0.f;
    for (int k = 0; k < KTOP; ++k) { w_s[k] = __expf(w_s[k] - mx); s += w_s[k]; }
    float inv = 1.0f / s;
    for (int k = 0; k < KTOP; ++k) w_s[k] *= inv;
  }
  __syncthreads();

  for (int d = tid; d < DD; d += 256) {
    float acc = 0.0f;
#pragma unroll
    for (int k = 0; k < KTOP; ++k) {
      int m = i_s[k];
      acc += w_s[k] * (features[(long long)mn_s[m] * DD + d] + pe_val(pos_s[m], d));
    }
    s_feat[(long long)n * DD + d] = acc;
  }
}

// ---------------- row softmax over attention scores ----------------
__global__ __launch_bounds__(256)
void softmax_rows(float* __restrict__ att, int rowlen)
{
  __shared__ float sred[256];
  const int tid = threadIdx.x;
  float* p = att + (long long)blockIdx.x * rowlen;

  float v[4];
  float mx = -3.4e38f;
#pragma unroll
  for (int i = 0; i < 4; ++i) { v[i] = p[tid + i * 256]; mx = fmaxf(mx, v[i]); }
  mx = block_max256(mx, sred);
  float s = 0.f;
#pragma unroll
  for (int i = 0; i < 4; ++i) { v[i] = __expf(v[i] - mx); s += v[i]; }
  s = block_sum256(s, sred);
  float inv = 1.0f / s;
#pragma unroll
  for (int i = 0; i < 4; ++i) p[tid + i * 256] = v[i] * inv;
}

// ---------------- fused residual add + LayerNorm ----------------
__global__ __launch_bounds__(256)
void add_ln_kernel(const float* __restrict__ a, const float* __restrict__ b,
                   const float* __restrict__ g, const float* __restrict__ bet,
                   float* __restrict__ o)
{
  __shared__ float sred[256];
  const int row = blockIdx.x;
  const int tid = threadIdx.x;
  float v[8];
  float s = 0.f;
#pragma unroll
  for (int i = 0; i < 8; ++i) {
    int d = tid + i * 256;
    v[i] = a[(long long)row * DD + d] + b[(long long)row * DD + d];
    s += v[i];
  }
  float mean = block_sum256(s, sred) * (1.0f / DD);
  float vs = 0.f;
#pragma unroll
  for (int i = 0; i < 8; ++i) { float d0 = v[i] - mean; vs += d0 * d0; }
  float var = block_sum256(vs, sred) * (1.0f / DD);
  float inv = rsqrtf(var + 1e-5f);
#pragma unroll
  for (int i = 0; i < 8; ++i) {
    int d = tid + i * 256;
    o[(long long)row * DD + d] = (v[i] - mean) * inv * g[d] + bet[d];
  }
}

// ---------------- BatchNorm over batch dim ----------------
__global__ __launch_bounds__(256)
void bn_stats_kernel(const float* __restrict__ h, float* __restrict__ mu,
                     float* __restrict__ var, int rows, int cols)
{
  int c = blockIdx.x * blockDim.x + threadIdx.x;
  if (c >= cols) return;
  float s = 0.f, sq = 0.f;
  for (int r = 0; r < rows; ++r) {
    float x = h[(long long)r * cols + c];
    s += x; sq += x * x;
  }
  float m = s / (float)rows;
  mu[c] = m;
  var[c] = sq / (float)rows - m * m;
}

__global__ __launch_bounds__(256)
void bn_apply_kernel(float* __restrict__ h, const float* __restrict__ mu,
                     const float* __restrict__ var, const float* __restrict__ g,
                     const float* __restrict__ b, int total, int cols)
{
  int i = blockIdx.x * blockDim.x + threadIdx.x;
  if (i >= total) return;
  int c = i % cols;
  float x = (h[i] - mu[c]) * rsqrtf(var[c] + 1e-5f) * g[c] + b[c];
  h[i] = fmaxf(x, 0.0f);
}

// ---------------- host side ----------------
static inline void launch_gemm(const float* A, long long lda, long long sA,
                               const float* B, long long ldb, long long sB,
                               float* C, long long ldc, long long sC,
                               int M_, int N_, int K_, const float* bias,
                               float alpha, int bT, int relu, int batch,
                               hipStream_t s)
{
  dim3 grid((N_ + BN - 1) / BN, (M_ + BM - 1) / BM, batch);
  gemm_bf16_wmma<<<grid, dim3(256), 0, s>>>(A, lda, sA, B, ldb, sB, C, ldc, sC,
                                            M_, N_, K_, bias, alpha, bT, relu);
}

extern "C" void kernel_launch(void* const* d_in, const int* in_sizes, int n_in,
                              void* d_out, int out_size, void* d_ws, size_t ws_size,
                              hipStream_t stream)
{
  (void)in_sizes; (void)n_in; (void)out_size; (void)ws_size;

  const float* features   = (const float*)d_in[0];
  const int*   frame_id   = (const int*)  d_in[1];
  const int*   match_node = (const int*)  d_in[2];

  const float* attw[2] = { (const float*)d_in[3],  (const float*)d_in[11] };
  const float* attb[2] = { (const float*)d_in[4],  (const float*)d_in[12] };
  const float* lng[2]  = { (const float*)d_in[5],  (const float*)d_in[13] };
  const float* lnb[2]  = { (const float*)d_in[6],  (const float*)d_in[14] };
  const float* fw1[2]  = { (const float*)d_in[7],  (const float*)d_in[15] };
  const float* fb1[2]  = { (const float*)d_in[8],  (const float*)d_in[16] };
  const float* fw2[2]  = { (const float*)d_in[9],  (const float*)d_in[17] };
  const float* fb2[2]  = { (const float*)d_in[10], (const float*)d_in[18] };

  const float* dec_w1 = (const float*)d_in[19];
  const float* dec_b1 = (const float*)d_in[20];
  const float* bn_g   = (const float*)d_in[21];
  const float* bn_b   = (const float*)d_in[22];
  const float* dec_w2 = (const float*)d_in[23];
  const float* dec_b2 = (const float*)d_in[24];

  float* out  = (float*)d_out;
  float* xout = out;                        // [N, D]
  float* dist = out + (size_t)NN * DD;      // [N, 35]

  float* w = (float*)d_ws;
  float* s_feat = w; w += (size_t)NN * DD;
  float* qb     = w; w += (size_t)NN * DD;
  float* kb     = w; w += (size_t)NN * DD;
  float* vb     = w; w += (size_t)NN * DD;
  float* att    = w; w += (size_t)HH * NN * NN;
  float* ao     = w; w += (size_t)NN * DD;
  float* pj     = w; w += (size_t)NN * DD;
  float* xA     = w; w += (size_t)NN * DD;
  float* xB     = w; w += (size_t)NN * DD;
  float* xM     = w; w += (size_t)NN * DD;
  float* fh     = w; w += (size_t)NN * DFFC;
  float* hd     = w; w += (size_t)NN * DEC1;
  float* mu     = w; w += DEC1;
  float* var    = w; w += DEC1;

  // ---- selector -> s_feat ----
  selector_kernel<<<NN, 256, 0, stream>>>(features, frame_id, match_node, s_feat);

  const float* x_cur = features;
  const float rscale = 0.0625f;             // 1/sqrt(dh=256)

  for (int enc = 0; enc < 2; ++enc) {
    for (int l = 0; l < LL; ++l) {
      const float* W  = attw[enc] + (size_t)l * 4 * DD * DD;
      const float* Bb = attb[enc] + (size_t)l * 4 * DD;
      const float* kv = (enc == 0) ? s_feat : x_cur;

      // q/k/v projections
      launch_gemm(x_cur, DD, 0, W + 0 * (size_t)DD * DD, DD, 0, qb, DD, 0,
                  NN, DD, DD, Bb + 0 * DD, 1.0f, 0, 0, 1, stream);
      launch_gemm(kv,    DD, 0, W + 1 * (size_t)DD * DD, DD, 0, kb, DD, 0,
                  NN, DD, DD, Bb + 1 * DD, 1.0f, 0, 0, 1, stream);
      launch_gemm(kv,    DD, 0, W + 2 * (size_t)DD * DD, DD, 0, vb, DD, 0,
                  NN, DD, DD, Bb + 2 * DD, 1.0f, 0, 0, 1, stream);

      // scores: att[h] = (q_h x k_h^T) / sqrt(dh)   (head-batched, B transposed)
      launch_gemm(qb, DD, DH, kb, DD, DH, att, NN, (long long)NN * NN,
                  NN, NN, DH, nullptr, rscale, 1, 0, HH, stream);

      softmax_rows<<<HH * NN, 256, 0, stream>>>(att, NN);

      // head outputs: o_h = att_h x v_h  (writes into [N, D] at col offset h*dh)
      launch_gemm(att, NN, (long long)NN * NN, vb, DD, DH, ao, DD, DH,
                  NN, DH, NN, nullptr, 1.0f, 0, 0, HH, stream);

      // output projection
      launch_gemm(ao, DD, 0, W + 3 * (size_t)DD * DD, DD, 0, pj, DD, 0,
                  NN, DD, DD, Bb + 3 * DD, 1.0f, 0, 0, 1, stream);

      // LN1(x + attn)
      add_ln_kernel<<<NN, 256, 0, stream>>>(
          x_cur, pj, lng[enc] + (size_t)(2 * l + 0) * DD,
          lnb[enc] + (size_t)(2 * l + 0) * DD, xM);

      // FFN
      launch_gemm(xM, DD, 0, fw1[enc] + (size_t)l * DD * DFFC, DFFC, 0, fh, DFFC, 0,
                  NN, DFFC, DD, fb1[enc] + (size_t)l * DFFC, 1.0f, 0, 1, 1, stream);
      launch_gemm(fh, DFFC, 0, fw2[enc] + (size_t)l * DFFC * DD, DD, 0, pj, DD, 0,
                  NN, DD, DFFC, fb2[enc] + (size_t)l * DD, 1.0f, 0, 0, 1, stream);

      // LN2(xM + ffn) -> next x (last spatial layer writes straight to d_out)
      float* nxt = (enc == 1 && l == LL - 1) ? xout : ((x_cur == xA) ? xB : xA);
      add_ln_kernel<<<NN, 256, 0, stream>>>(
          xM, pj, lng[enc] + (size_t)(2 * l + 1) * DD,
          lnb[enc] + (size_t)(2 * l + 1) * DD, nxt);
      x_cur = nxt;
    }
  }

  // ---- decoder ----
  launch_gemm(xout, DD, 0, dec_w1, DEC1, 0, hd, DEC1, 0,
              NN, DEC1, DD, dec_b1, 1.0f, 0, 0, 1, stream);
  bn_stats_kernel<<<(DEC1 + 255) / 256, 256, 0, stream>>>(hd, mu, var, NN, DEC1);
  bn_apply_kernel<<<(NN * DEC1 + 255) / 256, 256, 0, stream>>>(hd, mu, var, bn_g, bn_b,
                                                               NN * DEC1, DEC1);
  launch_gemm(hd, DEC1, 0, dec_w2, NC, 0, dist, NC, 0,
              NN, NC, DEC1, dec_b2, 1.0f, 0, 0, 1, stream);
}
